// GraphModel_58016418234712
// MI455X (gfx1250) — matrix operand order repaired and checked by reading, not compiled, not run on hardware
//
#include <hip/hip_runtime.h>

typedef __attribute__((ext_vector_type(16))) _Float16 v16h;
typedef __attribute__((ext_vector_type(8)))  float    v8f;

#define NN   200000
#define NE   6400000
#define CIN  128
#define HID  16
#define COUT 32

// ---------------- degree / dinv ----------------
__global__ void k_init_deg(float* deg) {
    int i = blockIdx.x * blockDim.x + threadIdx.x;
    if (i < NN) deg[i] = 1.0f;                      // self-loop
}
__global__ void k_deg_accum(const int* __restrict__ dst, float* deg) {
    int e = blockIdx.x * blockDim.x + threadIdx.x;
    if (e < NE) atomicAdd(&deg[dst[e]], 1.0f);
}
__global__ void k_deg_fin(float* deg) {
    int i = blockIdx.x * blockDim.x + threadIdx.x;
    if (i < NN) deg[i] = rsqrtf(deg[i]);            // deg >= 1 always
}

// ---------------- GEMM1: A1 = (x @ W1) * dinv[row], WMMA f16 16x16x32 ----------------
__global__ void k_gemm1(const float* __restrict__ x, const float* __restrict__ W1,
                        const float* __restrict__ dinv, float* __restrict__ A1) {
    const int lane = threadIdx.x & 31;
    const int hi   = lane >> 4;          // 0: lanes 0-15, 1: lanes 16-31
    const int m    = lane & 15;
    const int waveGlobal = blockIdx.x * (blockDim.x >> 5) + (threadIdx.x >> 5);
    const int numWaves   = gridDim.x * (blockDim.x >> 5);

    // B operand (W1, 128x16 row-major), loaded once per wave. B 32x16 k order used here (same as A):
    // lane N = l%16 ; half h -> K = kbase + 16*hi + h
    v16h b[4];
    #pragma unroll
    for (int c2 = 0; c2 < 4; ++c2) {
        const int kb = c2 * 32 + hi * 16;
        #pragma unroll
        for (int h = 0; h < 16; ++h)
            b[c2][h] = (_Float16)W1[(kb + h) * HID + m];
    }

    const int numTiles = NN / 16;        // 12500 exactly
    for (int tile = waveGlobal; tile < numTiles; tile += numWaves) {
        const int row = tile * 16 + m;   // A layout: lane M = l%16
        const float* xr = x + (size_t)row * CIN;
        v8f c = {0.f,0.f,0.f,0.f,0.f,0.f,0.f,0.f};
        #pragma unroll
        for (int c2 = 0; c2 < 4; ++c2) {
            // A 16x32 k order used here (same as B): halves 0-7 -> K = kbase + 16*hi + h
            //                 halves 8-15 -> K = kbase + 16*hi + 8 + h
            const int k1 = c2 * 32 + hi * 16;
            const int k2 = c2 * 32 + hi * 16 + 8;
            v16h a;
            #pragma unroll
            for (int h = 0; h < 8; ++h) {
                a[h]     = (_Float16)xr[k1 + h];
                a[h + 8] = (_Float16)xr[k2 + h];
            }
            c = __builtin_amdgcn_wmma_f32_16x16x32_f16(false, a, false, b[c2],
                                                       (short)0, c, false, false);
        }
        // C/D layout: lane l, vgpr v -> (M = v + 8*hi, Ncol = l%16)
        #pragma unroll
        for (int v = 0; v < 8; ++v) {
            const int mr = tile * 16 + v + hi * 8;
            A1[(size_t)mr * HID + m] = c[v] * dinv[mr];
        }
    }
}

// ---------------- generic float4 copy (self-loop term init) ----------------
__global__ void k_copy4(const float4* __restrict__ s, float4* __restrict__ d, int n4) {
    int i = blockIdx.x * blockDim.x + threadIdx.x;
    if (i < n4) d[i] = s[i];
}

// ---------------- edge scatter-add, CH channels/edge ----------------
template <int CH>
__global__ void k_scatter(const int* __restrict__ src, const int* __restrict__ dst,
                          const float* __restrict__ A, float* __restrict__ AGG) {
    int t = blockIdx.x * blockDim.x + threadIdx.x;
    if (t >= NE * CH) return;
    const int e  = t / CH;
    const int ch = t - e * CH;
    const int s = src[e], d = dst[e];
    atomicAdd(&AGG[(size_t)d * CH + ch], A[(size_t)s * CH + ch]);
}

// --------- GEMM2: B2 = (relu(AGG*dinv + b1) @ W2) * dinv, K=16 zero-padded WMMA ---------
__global__ void k_gemm2(const float* __restrict__ AGG, const float* __restrict__ W2,
                        const float* __restrict__ b1, const float* __restrict__ dinv,
                        float* __restrict__ B2) {
    const int lane = threadIdx.x & 31;
    const int hi   = lane >> 4;
    const int m    = lane & 15;
    const int waveGlobal = blockIdx.x * (blockDim.x >> 5) + (threadIdx.x >> 5);
    const int numWaves   = gridDim.x * (blockDim.x >> 5);

    // B operand (W2, 16x32 row-major): halves 0-7 carry K=8*hi+h (0..15), halves 8-15 carry K=16..31 -> 0
    v16h b[2];
    #pragma unroll
    for (int nb = 0; nb < 2; ++nb)
        #pragma unroll
        for (int h = 0; h < 16; ++h)
            b[nb][h] = (h >= 8) ? (_Float16)0.0f : (_Float16)W2[(hi * 8 + h) * COUT + nb * 16 + m];

    float bias[8];
    const int koff = hi * 8;             // lanes 0-15: K 0..7 ; lanes 16-31: K 8..15
    #pragma unroll
    for (int h = 0; h < 8; ++h) bias[h] = b1[koff + h];

    const int numTiles = NN / 16;
    for (int tile = waveGlobal; tile < numTiles; tile += numWaves) {
        const int row = tile * 16 + m;
        const float di = dinv[row];
        const float* ar = AGG + (size_t)row * HID + koff;
        v16h a;
        #pragma unroll
        for (int h = 0; h < 8; ++h) {
            const float v = fmaf(ar[h], di, bias[h]);      // layer-1 post-scale + bias
            a[h]     = (_Float16)fmaxf(v, 0.0f);           // relu, fused
            a[h + 8] = (_Float16)0.0f;                     // K = 16..31 padding
        }
        float dv[8];
        #pragma unroll
        for (int v = 0; v < 8; ++v) dv[v] = dinv[tile * 16 + v + hi * 8];
        #pragma unroll
        for (int nb = 0; nb < 2; ++nb) {
            v8f c = {0.f,0.f,0.f,0.f,0.f,0.f,0.f,0.f};
            c = __builtin_amdgcn_wmma_f32_16x16x32_f16(false, a, false, b[nb],
                                                       (short)0, c, false, false);
            #pragma unroll
            for (int v = 0; v < 8; ++v) {
                const int mr = tile * 16 + v + hi * 8;
                B2[(size_t)mr * COUT + nb * 16 + m] = c[v] * dv[v];
            }
        }
    }
}

// ---------------- finalize: out = out * dinv + b2 (in place) ----------------
__global__ void k_finalize(float* __restrict__ out, const float* __restrict__ dinv,
                           const float* __restrict__ b2) {
    int t = blockIdx.x * blockDim.x + threadIdx.x;
    if (t >= NN * COUT) return;
    const int i  = t / COUT;
    const int ch = t - i * COUT;
    out[t] = fmaf(out[t], dinv[i], b2[ch]);
}

extern "C" void kernel_launch(void* const* d_in, const int* in_sizes, int n_in,
                              void* d_out, int out_size, void* d_ws, size_t ws_size,
                              hipStream_t stream) {
    const float* x  = (const float*)d_in[0];
    const int*   ei = (const int*)d_in[1];       // [2, E] flattened
    const float* W1 = (const float*)d_in[2];
    const float* b1 = (const float*)d_in[3];
    const float* W2 = (const float*)d_in[4];
    const float* b2 = (const float*)d_in[5];
    const int* src = ei;
    const int* dst = ei + NE;
    float* out = (float*)d_out;

    // workspace layout (floats), 1024-aligned regions
    float* ws   = (float*)d_ws;
    float* dinv = ws;                         // NN          (deg, then rsqrt in place)
    float* A1   = ws + 200704;                // NN*16  prescaled layer-1 GEMM out
    float* AGG  = A1 + 3200000;               // NN*16  layer-1 aggregate
    float* B2   = AGG + 3200000;              // NN*32  prescaled layer-2 GEMM out

    const int T = 256;
    // degrees -> dinv
    k_init_deg <<<(NN + T - 1) / T, T, 0, stream>>>(dinv);
    k_deg_accum<<<(NE + T - 1) / T, T, 0, stream>>>(dst, dinv);
    k_deg_fin  <<<(NN + T - 1) / T, T, 0, stream>>>(dinv);
    // layer 1: GEMM (WMMA) with source pre-scale
    k_gemm1<<<512, T, 0, stream>>>(x, W1, dinv, A1);
    // aggregate 1: self loop + edges
    k_copy4<<<(NN * HID / 4 + T - 1) / T, T, 0, stream>>>((const float4*)A1, (float4*)AGG, NN * HID / 4);
    k_scatter<HID><<<(NE * HID + T - 1) / T, T, 0, stream>>>(src, dst, A1, AGG);
    // layer 2: fused relu/bias/post-scale + GEMM (WMMA) with source pre-scale
    k_gemm2<<<512, T, 0, stream>>>(AGG, W2, b1, dinv, B2);
    // aggregate 2: self loop + edges, straight into d_out
    k_copy4<<<(NN * COUT / 4 + T - 1) / T, T, 0, stream>>>((const float4*)B2, (float4*)out, NN * COUT / 4);
    k_scatter<COUT><<<(NE * COUT + T - 1) / T, T, 0, stream>>>(src, dst, B2, out);
    // out = out * dinv + b2
    k_finalize<<<(NN * COUT + T - 1) / T, T, 0, stream>>>(out, dinv, b2);
}
